// Attention_28991029248800
// MI455X (gfx1250) — matrix operand order repaired and checked by reading, not compiled, not run on hardware
//
#include <hip/hip_runtime.h>

typedef __attribute__((ext_vector_type(16))) __bf16 v16bf;
typedef __attribute__((ext_vector_type(8)))  __bf16 v8bf;
typedef __attribute__((ext_vector_type(8)))  float  v8f;

constexpr int SEQ  = 2048;
constexpr int DIM  = 4096;
constexpr int HD   = 128;
constexpr int NH   = 32;
constexpr int NKV  = 8;
constexpr int KVD  = NKV * HD;      // 1024
constexpr int NTOT = DIM + 2 * KVD; // 6144 fused QKV output columns

__device__ __forceinline__ v8f wmma_bf16(v16bf a, v16bf b, v8f c) {
  return __builtin_amdgcn_wmma_f32_16x16x32_bf16(false, a, false, b, (short)0, c,
                                                 false, false);
}

// A-fragment loader for bf16 memory: lane holds row M, elements 0..15 at
// K=kb..kb+15 (kb = half*16 folded into the pointer).
__device__ __forceinline__ v16bf ldA_bf(const __bf16* p) {
  v8bf lo = *(const v8bf*)(p);
  v8bf hi = *(const v8bf*)(p + 8);
  v16bf r;
#pragma unroll
  for (int i = 0; i < 8; ++i) { r[i] = lo[i]; r[i + 8] = hi[i]; }
  return r;
}

// ---------------------------------------------------------------------------
// Kernel 0: one-shot fp32 -> bf16 conversion of x, [wq;wk;wv], wo.
// Each thread converts 8 contiguous floats (two b128 loads -> one b128 store).
// ---------------------------------------------------------------------------
__global__ __launch_bounds__(256) void conv_kernel(
    const float* __restrict__ x, const float* __restrict__ wq,
    const float* __restrict__ wk, const float* __restrict__ wv,
    const float* __restrict__ wo, __bf16* __restrict__ xb,
    __bf16* __restrict__ wqkvb, __bf16* __restrict__ wob) {
  constexpr size_t NX = (size_t)SEQ * DIM;   //  8,388,608
  constexpr size_t NQ = (size_t)DIM * DIM;   // 16,777,216
  constexpr size_t NK = (size_t)KVD * DIM;   //  4,194,304
  constexpr size_t TOT = NX + NQ + 2 * NK + NQ;
  size_t idx = ((size_t)blockIdx.x * blockDim.x + threadIdx.x) * 8;
  if (idx >= TOT) return;
  const float* src;
  __bf16* dst;
  size_t off;
  if (idx < NX)                { src = x;  dst = xb;              off = idx; }
  else if (idx < NX + NQ)      { src = wq; dst = wqkvb;           off = idx - NX; }
  else if (idx < NX + NQ + NK) { src = wk; dst = wqkvb + NQ;      off = idx - NX - NQ; }
  else if (idx < NX + NQ + 2 * NK) {
    src = wv; dst = wqkvb + NQ + NK; off = idx - NX - NQ - NK;
  } else                       { src = wo; dst = wob;             off = idx - NX - NQ - 2 * NK; }
  float4 a = *(const float4*)(src + off);
  float4 b = *(const float4*)(src + off + 4);
  v8bf r;
  r[0] = (__bf16)a.x; r[1] = (__bf16)a.y; r[2] = (__bf16)a.z; r[3] = (__bf16)a.w;
  r[4] = (__bf16)b.x; r[5] = (__bf16)b.y; r[6] = (__bf16)b.z; r[7] = (__bf16)b.w;
  *(v8bf*)(dst + off) = r;
}

// ---------------------------------------------------------------------------
// Kernel 1: fused QKV projection (pure bf16 operands), 32x64 tile per wave:
// two A fragments share each B fragment (2x arithmetic intensity vs 16x64).
// C[s,o] = sum_d xb[s,d] * Wb[o,d], o in [0,6144).
// Q -> qbuf[h][s][128], K -> kbuf[h][s][128], V -> vtbuf[h][d][s] (transposed).
// ---------------------------------------------------------------------------
__global__ __launch_bounds__(256) void qkv_kernel(
    const __bf16* __restrict__ xb, const __bf16* __restrict__ wqkvb,
    __bf16* __restrict__ qbuf, __bf16* __restrict__ kbuf,
    __bf16* __restrict__ vtbuf) {
  const int lane = threadIdx.x & 31;
  const int wid  = threadIdx.x >> 5;
  const int half = lane >> 4;
  const int l16  = lane & 15;
  const int mbase = blockIdx.x * 32;
  const int nbase = blockIdx.y * 512 + wid * 64;  // 64 output cols per wave

  v8f acc[2][4] = {};
  const __bf16* arow0 = xb + (size_t)(mbase + l16) * DIM;
  const __bf16* arow1 = xb + (size_t)(mbase + 16 + l16) * DIM;

  for (int ks = 0; ks < DIM / 32; ++ks) {
    v16bf af0 = ldA_bf(arow0 + ks * 32 + half * 16);
    v16bf af1 = ldA_bf(arow1 + ks * 32 + half * 16);
    const int kbb = ks * 32 + half * 16;
#pragma unroll
    for (int nt = 0; nt < 4; ++nt) {
      v16bf bf =
          *(const v16bf*)(wqkvb + (size_t)(nbase + nt * 16 + l16) * DIM + kbb);
      acc[0][nt] = wmma_bf16(af0, bf, acc[0][nt]);
      acc[1][nt] = wmma_bf16(af1, bf, acc[1][nt]);
    }
  }

#pragma unroll
  for (int mt = 0; mt < 2; ++mt) {
#pragma unroll
    for (int nt = 0; nt < 4; ++nt) {
      const int gcol = nbase + nt * 16 + l16;
#pragma unroll
      for (int r = 0; r < 8; ++r) {
        const int srow = mbase + mt * 16 + r + half * 8;
        __bf16 bv = (__bf16)acc[mt][nt][r];
        if (gcol < DIM) {
          const int h = gcol >> 7, d = gcol & 127;
          qbuf[((size_t)h * SEQ + srow) * HD + d] = bv;
        } else if (gcol < DIM + KVD) {
          const int c = gcol - DIM, h = c >> 7, d = c & 127;
          kbuf[((size_t)h * SEQ + srow) * HD + d] = bv;
        } else {
          const int c = gcol - DIM - KVD, h = c >> 7, d = c & 127;
          vtbuf[((size_t)h * HD + d) * SEQ + srow] = bv;  // transposed V
        }
      }
    }
  }
}

// ---------------------------------------------------------------------------
// Kernel 2: RoPE on Q and K (pairwise rotate in fp32, restore bf16).
// ---------------------------------------------------------------------------
__global__ void rope_kernel(__bf16* __restrict__ qbuf, __bf16* __restrict__ kbuf,
                            const float* __restrict__ fc,
                            const float* __restrict__ fs) {
  const int NQP = NH * SEQ * (HD / 2);
  const int NKP = NKV * SEQ * (HD / 2);
  int idx = blockIdx.x * blockDim.x + threadIdx.x;
  if (idx >= NQP + NKP) return;
  __bf16* buf;
  int rem;
  if (idx < NQP) { buf = qbuf; rem = idx; }
  else           { buf = kbuf; rem = idx - NQP; }
  const int h = rem / (SEQ * 64);
  rem %= SEQ * 64;
  const int s = rem / 64, i = rem % 64;
  __bf16* p = buf + ((size_t)h * SEQ + s) * HD + 2 * i;
  const float a = (float)p[0], b = (float)p[1];
  const float c = fc[s * 64 + i], sn = fs[s * 64 + i];
  p[0] = (__bf16)(a * c - b * sn);
  p[1] = (__bf16)(a * sn + b * c);
}

// ---------------------------------------------------------------------------
// Kernel 3: causal flash attention. One wave per (head, 16-query tile).
// Key blocks of 32: 8 score WMMAs, online softmax, 8 P*V WMMAs into f32 O.
// Q fragments are loop-invariant and hoisted out of the key loop.
// ---------------------------------------------------------------------------
__global__ __launch_bounds__(256) void attn_kernel(
    const __bf16* __restrict__ qbuf, const __bf16* __restrict__ kbuf,
    const __bf16* __restrict__ vtbuf, __bf16* __restrict__ attn) {
  __shared__ __align__(32) __bf16 pshared[8 * 16 * 32];  // per-wave 16x32 P tile
  const int lane = threadIdx.x & 31;
  const int wid  = threadIdx.x >> 5;
  const int half = lane >> 4;
  const int l16  = lane & 15;
  const int g = blockIdx.x * 8 + wid;
  const int qt = g & 127;      // 128 query tiles
  const int h  = g >> 7;       // 32 heads
  const int kvh = h >> 2;      // N_REP = 4
  const int qbase = qt * 16;
  __bf16* pb = pshared + wid * 512;

  const __bf16* qrow = qbuf + ((size_t)h * SEQ + (qbase + l16)) * HD;
  const __bf16* kbp  = kbuf + (size_t)kvh * SEQ * HD;
  const __bf16* vtp  = vtbuf + (size_t)kvh * HD * SEQ;

  v16bf qf[4];
#pragma unroll
  for (int ks = 0; ks < 4; ++ks) qf[ks] = ldA_bf(qrow + ks * 32 + half * 16);

  float mi[8], li[8], alpha[8];
  v8f o[8] = {};
#pragma unroll
  for (int r = 0; r < 8; ++r) { mi[r] = -1e30f; li[r] = 0.f; }
  const float rscale = 0.08838834764831845f;  // 1/sqrt(128)

  const int nblk = (qbase + 15) / 32 + 1;
  for (int kb = 0; kb < nblk; ++kb) {
    const int keybase = kb * 32;
    v8f s0 = {}, s1 = {};
#pragma unroll
    for (int ks = 0; ks < 4; ++ks) {
      const int kofs = ks * 32 + half * 16;
      v16bf b0 = *(const v16bf*)(kbp + (size_t)(keybase + l16) * HD + kofs);
      v16bf b1 = *(const v16bf*)(kbp + (size_t)(keybase + 16 + l16) * HD + kofs);
      s0 = wmma_bf16(qf[ks], b0, s0);
      s1 = wmma_bf16(qf[ks], b1, s1);
    }
    const bool need_mask = (keybase + 31) > qbase;
#pragma unroll
    for (int r = 0; r < 8; ++r) {
      const int qr = qbase + r + half * 8;
      float v0 = s0[r] * rscale;
      float v1 = s1[r] * rscale;
      if (need_mask) {
        if (keybase + l16 > qr)      v0 = -1e9f;
        if (keybase + 16 + l16 > qr) v1 = -1e9f;
      }
      float mx = fmaxf(v0, v1);
      mx = fmaxf(mx, __shfl_xor(mx, 1, 32));
      mx = fmaxf(mx, __shfl_xor(mx, 2, 32));
      mx = fmaxf(mx, __shfl_xor(mx, 4, 32));
      mx = fmaxf(mx, __shfl_xor(mx, 8, 32));
      const float mn = fmaxf(mi[r], mx);
      const float p0 = __expf(v0 - mn);
      const float p1 = __expf(v1 - mn);
      float rs = p0 + p1;
      rs += __shfl_xor(rs, 1, 32);
      rs += __shfl_xor(rs, 2, 32);
      rs += __shfl_xor(rs, 4, 32);
      rs += __shfl_xor(rs, 8, 32);
      const float al = __expf(mi[r] - mn);
      li[r] = li[r] * al + rs;
      mi[r] = mn;
      alpha[r] = al;
      const int prow = r + half * 8;
      pb[prow * 32 + l16]      = (__bf16)p0;
      pb[prow * 32 + 16 + l16] = (__bf16)p1;
    }
    // Same-wave LDS store->load of the P tile: LDS is in-order per wave; fence
    // the compiler and drain DScnt before re-reading in A-fragment layout.
    asm volatile("s_wait_dscnt 0" ::: "memory");
#pragma unroll
    for (int t = 0; t < 8; ++t)
#pragma unroll
      for (int r = 0; r < 8; ++r) o[t][r] *= alpha[r];
    v16bf pf = ldA_bf(pb + l16 * 32 + half * 16);
#pragma unroll
    for (int t = 0; t < 8; ++t) {
      v16bf vf =
          *(const v16bf*)(vtp + (size_t)(t * 16 + l16) * SEQ + keybase + half * 16);
      o[t] = wmma_bf16(pf, vf, o[t]);
    }
  }

#pragma unroll
  for (int t = 0; t < 8; ++t) {
    const int gcol = h * HD + t * 16 + l16;
#pragma unroll
    for (int r = 0; r < 8; ++r) {
      const int srow = qbase + r + half * 8;
      attn[(size_t)srow * DIM + gcol] = (__bf16)(o[t][r] / li[r]);
    }
  }
}

// ---------------------------------------------------------------------------
// Kernel 4: output projection, 32x64 tile per wave (same blocking as QKV).
// out[s,d] = sum_o attn[s,o] * wob[d,o] (fp32 out).
// ---------------------------------------------------------------------------
__global__ __launch_bounds__(256) void oproj_kernel(
    const __bf16* __restrict__ attn, const __bf16* __restrict__ wob,
    float* __restrict__ out) {
  const int lane = threadIdx.x & 31;
  const int wid  = threadIdx.x >> 5;
  const int half = lane >> 4;
  const int l16  = lane & 15;
  const int mbase = blockIdx.x * 32;
  const int nbase = blockIdx.y * 512 + wid * 64;

  v8f acc[2][4] = {};
  const __bf16* arow0 = attn + (size_t)(mbase + l16) * DIM;
  const __bf16* arow1 = attn + (size_t)(mbase + 16 + l16) * DIM;

  for (int ks = 0; ks < DIM / 32; ++ks) {
    v16bf af0 = ldA_bf(arow0 + ks * 32 + half * 16);
    v16bf af1 = ldA_bf(arow1 + ks * 32 + half * 16);
    const int kbb = ks * 32 + half * 16;
#pragma unroll
    for (int nt = 0; nt < 4; ++nt) {
      v16bf bf =
          *(const v16bf*)(wob + (size_t)(nbase + nt * 16 + l16) * DIM + kbb);
      acc[0][nt] = wmma_bf16(af0, bf, acc[0][nt]);
      acc[1][nt] = wmma_bf16(af1, bf, acc[1][nt]);
    }
  }
#pragma unroll
  for (int mt = 0; mt < 2; ++mt) {
#pragma unroll
    for (int nt = 0; nt < 4; ++nt) {
      const int gcol = nbase + nt * 16 + l16;
#pragma unroll
      for (int r = 0; r < 8; ++r) {
        const int srow = mbase + mt * 16 + r + half * 8;
        out[(size_t)srow * DIM + gcol] = acc[mt][nt][r];
      }
    }
  }
}

// ---------------------------------------------------------------------------
extern "C" void kernel_launch(void* const* d_in, const int* in_sizes, int n_in,
                              void* d_out, int out_size, void* d_ws,
                              size_t ws_size, hipStream_t stream) {
  const float* x  = (const float*)d_in[0];
  const float* wq = (const float*)d_in[1];
  const float* wk = (const float*)d_in[2];
  const float* wv = (const float*)d_in[3];
  const float* wo = (const float*)d_in[4];
  const float* fc = (const float*)d_in[5];
  const float* fs = (const float*)d_in[6];
  float* out = (float*)d_out;

  // Workspace layout (bytes). attn aliases xb: xb is dead after qkv_kernel.
  char* ws = (char*)d_ws;
  __bf16* xb    = (__bf16*)(ws);                        // 16 MiB (bf16 x)
  __bf16* attn  = (__bf16*)(ws);                        // 16 MiB (aliases xb)
  __bf16* wqkvb = (__bf16*)(ws + (size_t)16777216);     // 48 MiB (bf16 wq|wk|wv)
  __bf16* wob   = (__bf16*)(ws + (size_t)67108864);     // 32 MiB (bf16 wo)
  __bf16* qbuf  = (__bf16*)(ws + (size_t)100663296);    // 16 MiB
  __bf16* kbuf  = (__bf16*)(ws + (size_t)117440512);    //  4 MiB
  __bf16* vtbuf = (__bf16*)(ws + (size_t)121634816);    //  4 MiB (transposed V)

  dim3 blk(256);
  const size_t nconv =
      ((size_t)SEQ * DIM + 2 * (size_t)DIM * DIM + 2 * (size_t)KVD * DIM) / 8;
  conv_kernel<<<(unsigned)((nconv + 255) / 256), blk, 0, stream>>>(
      x, wq, wk, wv, wo, xb, wqkvb, wob);
  qkv_kernel<<<dim3(64, NTOT / 512), blk, 0, stream>>>(xb, wqkvb, qbuf, kbuf,
                                                       vtbuf);
  const int npairs = (NH + NKV) * SEQ * (HD / 2);
  rope_kernel<<<(npairs + 255) / 256, blk, 0, stream>>>(qbuf, kbuf, fc, fs);
  attn_kernel<<<(NH * 128) / 8, blk, 0, stream>>>(qbuf, kbuf, vtbuf, attn);
  oproj_kernel<<<dim3(64, DIM / 512), blk, 0, stream>>>(attn, wob, out);
}
